// EMLTokenPool_3332894621840
// MI455X (gfx1250) — matrix operand order repaired and checked by reading, not compiled, not run on hardware
//
#include <hip/hip_runtime.h>
#include <hip/hip_bf16.h>
#include <math.h>

// Problem constants (from reference setup_inputs)
#define BB 32
#define NN 4096
#define DD 512
#define HH 128
#define TOK (BB*NN)          // 131072 tokens
#define LN_EPS_F 1e-5f

// WMMA vector types (probe-confirmed shapes)
typedef __attribute__((ext_vector_type(16))) __bf16 v16bf;
typedef __attribute__((ext_vector_type(8)))  float  v8f;
typedef __attribute__((ext_vector_type(2)))  float  v2f;
typedef __attribute__((ext_vector_type(4)))  unsigned v4u;
typedef __attribute__((ext_vector_type(8)))  unsigned v8u;

// ---------------- workspace layout (bytes) ----------------
#define NRM_ELEMS   ((long long)TOK*DD)
#define W1F_ELEMS   (16*16*32*16)     // 131072 bf16 (= 4 phases x 64KB)
#define VWF_ELEMS   (32*16*32*16)     // 262144 bf16
#define W1F_OFF_B   (NRM_ELEMS*2)
#define VWF_OFF_B   (W1F_OFF_B + (long long)W1F_ELEMS*2)
#define PART_OFF_B  (VWF_OFF_B + (long long)VWF_ELEMS*2)
// out layout (floats): pooled[16384] weights[131072] energy[131072] drive[131072] res[131072]
#define OUT_W_OFF   16384
#define OUT_E_OFF   (16384+131072)
#define OUT_D_OFF   (16384+2*131072)
#define OUT_R_OFF   (16384+3*131072)

__device__ __forceinline__ unsigned short f2bf(float f) {
  unsigned u = __builtin_bit_cast(unsigned, f);
  unsigned r = u + 0x7fffu + ((u >> 16) & 1u);      // round-to-nearest-even
  return (unsigned short)(r >> 16);
}
__device__ __forceinline__ float bf2f(unsigned short h) {
  unsigned u = ((unsigned)h) << 16;
  return __builtin_bit_cast(float, u);
}
struct BFPack { uint4 a, b; };
__device__ __forceinline__ v16bf mk16(uint4 a, uint4 b) {
  BFPack p{a, b};
  return __builtin_bit_cast(v16bf, p);
}

// ---------------------------------------------------------------------------
// TDM: 1-D DMA of 64KB (8192 x 8B elements) global -> LDS.
// D# group0: count=1 | lds_addr | global_addr(57b) | type=2
// D# group1: data_size=3 (8B), tensor_dim0=tile_dim0=8192, dims1..=1/0,
//            tensor_dim0_stride=8192, no pad/iterate/multicast.
// Issued by one wave; tracked with TENSORcnt (in-order per wave).
// ---------------------------------------------------------------------------
__device__ __forceinline__ void tdm_load_64k(const void* gsrc, unsigned lds_off) {
  unsigned long long ga = (unsigned long long)(uintptr_t)gsrc;
  v4u g0;
  g0[0] = 1u;                                           // count=1, user mode
  g0[1] = lds_off;                                      // lds_addr (bytes)
  g0[2] = (unsigned)ga;                                 // global_addr[31:0]
  g0[3] = (unsigned)((ga >> 32) & 0x1FFFFFFull) | (2u << 30);  // addr[56:32] | type=2
  v8u g1;
  g1[0] = (3u << 16);          // workgroup_mask=0, data_size=3 (8B)
  g1[1] = (0x2000u << 16);     // tensor_dim0[15:0]=8192
  g1[2] = (1u << 16);          // tensor_dim0[31:16]=0 | tensor_dim1[15:0]=1
  g1[3] = (0x2000u << 16);     // tensor_dim1[31:16]=0 | tile_dim0=8192
  g1[4] = 1u;                  // tile_dim1=1, tile_dim2=0
  g1[5] = 0x2000u;             // tensor_dim0_stride[31:0]=8192
  g1[6] = 0u;                  // stride[47:32]=0 | tensor_dim1_stride[15:0]=0
  g1[7] = 0u;
  asm volatile("tensor_load_to_lds %0, %1" :: "s"(g0), "s"(g1) : "memory");
}

// ---------------------------------------------------------------------------
// k0: swizzle weights into per-lane-contiguous bf16 B-fragments.
// B fragment k order used here (A matches): lane L holds column N = ntile*16 + (L&15),
// K = kstep*32 + (L>>4)*16 + e, e=0..15 ascending (packed pairs per VGPR).
// ---------------------------------------------------------------------------
__global__ void k0_prep(const float* __restrict__ dw1, const float* __restrict__ rw1,
                        const float* __restrict__ vw,
                        unsigned short* __restrict__ w1f,
                        unsigned short* __restrict__ vwf) {
  int i = blockIdx.x * 256 + threadIdx.x;            // 0 .. 262143
  {
    int e = i & 15, lane = (i >> 4) & 31, ks = (i >> 9) & 15, nt = i >> 13; // nt 0..31
    int K   = ks * 32 + (lane >> 4) * 16 + e;
    int col = nt * 16 + (lane & 15);
    vwf[i] = f2bf(vw[K * DD + col]);
  }
  if (i < W1F_ELEMS) {
    int e = i & 15, lane = (i >> 4) & 31, ks = (i >> 9) & 15, nt = i >> 13; // nt 0..15
    int K   = ks * 32 + (lane >> 4) * 16 + e;
    int col = nt * 16 + (lane & 15);
    float v = (col < HH) ? dw1[K * HH + col] : rw1[K * HH + (col - HH)];
    w1f[i] = f2bf(v);
  }
}

// ---------------------------------------------------------------------------
// k1: LayerNorm + dual MLP (WMMA bf16, B double-buffer-staged in LDS via TDM)
// + EML energy.  64 tokens / block, 8 waves. 4 phases x 4 N-tiles
// (phases 0-1: drive MLP, phases 2-3: resistance MLP).
// Layer-2 dot fused into the WMMA epilogue.
// ---------------------------------------------------------------------------
#define NRM_STRIDE 520   // 512 + 8 bf16 pad -> staggered banks for b128 A reads

__global__ void __launch_bounds__(256)
k1_ln_mlp_energy(const float* __restrict__ tokens,
                 const float* __restrict__ ln_g, const float* __restrict__ ln_b,
                 const float* __restrict__ db1, const float* __restrict__ dw2,
                 const float* __restrict__ db2,
                 const float* __restrict__ rb1, const float* __restrict__ rw2,
                 const float* __restrict__ rb2,
                 const float* __restrict__ eml_gamma, const float* __restrict__ eml_lambda,
                 const float* __restrict__ eml_bias,
                 const unsigned short* __restrict__ w1frag,
                 unsigned short* __restrict__ nrm_out,
                 float* __restrict__ out_energy,
                 float* __restrict__ out_drive,
                 float* __restrict__ out_res) {
  __shared__ unsigned short nrm_lds[64 * NRM_STRIDE];   // 66.5 KB
  __shared__ unsigned short bfrag[2][32768];            // 2 x 64 KB B-fragment buffers
  __shared__ float w2_lds[256];
  __shared__ float b1_lds[256];
  __shared__ float drp[2][64][2];                       // [drive/res][token][wavegroup]

  const int tid  = threadIdx.x;
  const int wave = tid >> 5;
  const int lane = tid & 31;
  const int l16  = lane & 15, hl = lane >> 4;
  const long long tokBase = (long long)blockIdx.x * 64;
  const unsigned bf_off0 = (unsigned)(uintptr_t)(&bfrag[0][0]);
  const unsigned bf_off1 = (unsigned)(uintptr_t)(&bfrag[1][0]);

  // kick off phase-0/1 B-fragment DMAs early (overlap with LayerNorm)
  if (wave == 0) {
    tdm_load_64k(w1frag, bf_off0);
    tdm_load_64k((const char*)w1frag + 65536, bf_off1);
  }

  if (tid < 128) {
    w2_lds[tid]       = dw2[tid];
    w2_lds[128 + tid] = rw2[tid];
    b1_lds[tid]       = db1[tid];
    b1_lds[128 + tid] = rb1[tid];
  }

  // ---- LayerNorm: each wave handles 8 tokens; lane covers 16 columns ----
  float g[16], bta[16];
  #pragma unroll
  for (int j = 0; j < 4; ++j) {
    float4 gv = ((const float4*)ln_g)[lane * 4 + j];
    float4 bv = ((const float4*)ln_b)[lane * 4 + j];
    g[4*j+0] = gv.x; g[4*j+1] = gv.y; g[4*j+2] = gv.z; g[4*j+3] = gv.w;
    bta[4*j+0] = bv.x; bta[4*j+1] = bv.y; bta[4*j+2] = bv.z; bta[4*j+3] = bv.w;
  }

  #pragma unroll
  for (int i = 0; i < 8; ++i) {
    const int t = wave * 8 + i;
    const long long row = tokBase + t;
    const float4* src = (const float4*)(tokens + row * DD) + lane * 4;
    float x[16], s = 0.f, s2 = 0.f;
    #pragma unroll
    for (int j = 0; j < 4; ++j) {
      float4 v = src[j];
      x[4*j+0] = v.x; x[4*j+1] = v.y; x[4*j+2] = v.z; x[4*j+3] = v.w;
      s  += v.x + v.y + v.z + v.w;
      s2 += v.x*v.x + v.y*v.y + v.z*v.z + v.w*v.w;
    }
    #pragma unroll
    for (int m = 16; m >= 1; m >>= 1) {
      s  += __shfl_xor(s,  m, 32);
      s2 += __shfl_xor(s2, m, 32);
    }
    const float mu   = s * (1.0f / DD);
    const float var  = s2 * (1.0f / DD) - mu * mu;
    const float rstd = rsqrtf(var + LN_EPS_F);

    unsigned pk[8];
    #pragma unroll
    for (int j = 0; j < 8; ++j) {
      float n0 = (x[2*j+0] - mu) * rstd * g[2*j+0] + bta[2*j+0];
      float n1 = (x[2*j+1] - mu) * rstd * g[2*j+1] + bta[2*j+1];
      pk[j] = (unsigned)f2bf(n0) | ((unsigned)f2bf(n1) << 16);
    }
    uint4 p0 = make_uint4(pk[0], pk[1], pk[2], pk[3]);
    uint4 p1 = make_uint4(pk[4], pk[5], pk[6], pk[7]);
    unsigned short* ldst = &nrm_lds[t * NRM_STRIDE + lane * 16];
    *(uint4*)(ldst)      = p0;
    *(uint4*)(ldst + 8)  = p1;
    unsigned short* gdst = nrm_out + row * DD + lane * 16;
    *(uint4*)(gdst)      = p0;
    *(uint4*)(gdst + 8)  = p1;
  }
  __syncthreads();   // nrm_lds ready

  // ---- 4 software-pipelined phases of WMMA over TDM-staged B fragments ----
  // wave -> (M-tile = wave&3, group g = wave>>2); per phase a wave computes
  // 2 N-tiles: local idx {g*2, g*2+1} of the 4 staged tiles.
  const int mt = wave & 3;
  const int gp = wave >> 2;
  const int rowA = mt * 16 + l16;
  const int aoff = hl * 16;
  float pd[8], pr[8];
  #pragma unroll
  for (int r = 0; r < 8; ++r) { pd[r] = 0.f; pr[r] = 0.f; }

  #pragma unroll
  for (int p = 0; p < 4; ++p) {
    // loads complete in issue order per wave: <=1 outstanding => load p done,
    // load p+1 may still be in flight (double buffering).
    // builtin requires a literal immediate -> two call sites.
    if (p < 3) __builtin_amdgcn_s_wait_tensorcnt(1);
    else       __builtin_amdgcn_s_wait_tensorcnt(0);
    __syncthreads();                        // staged buffer visible to all waves

    const unsigned short* buf = &bfrag[p & 1][0];
    v8f acc0 = (v8f){0,0,0,0,0,0,0,0};
    v8f acc1 = (v8f){0,0,0,0,0,0,0,0};
    const int li0 = gp * 2, li1 = gp * 2 + 1;
    for (int ks = 0; ks < 16; ++ks) {
      const unsigned short* ap = &nrm_lds[rowA * NRM_STRIDE + ks * 32 + aoff];
      v16bf Af = mk16(*(const uint4*)(ap), *(const uint4*)(ap + 8));
      const uint4* b0 = (const uint4*)(buf + ((li0 * 16 + ks) * 32 + lane) * 16);
      const uint4* b1 = (const uint4*)(buf + ((li1 * 16 + ks) * 32 + lane) * 16);
      v16bf Bf0 = mk16(b0[0], b0[1]);
      v16bf Bf1 = mk16(b1[0], b1[1]);
      acc0 = __builtin_amdgcn_wmma_f32_16x16x32_bf16(
          false, Af, false, Bf0, (short)0, acc0, false, false);
      acc1 = __builtin_amdgcn_wmma_f32_16x16x32_bf16(
          false, Af, false, Bf1, (short)0, acc1, false, false);
    }
    __syncthreads();                        // all waves done reading buf[p&1]
    if (p < 2 && wave == 0)                 // refill freed buffer with load p+2
      tdm_load_64k((const char*)w1frag + (p + 2) * 65536,
                   (p & 1) ? bf_off1 : bf_off0);

    // epilogue: bias + exact GELU, fuse layer-2 dot (col contribution)
    #pragma unroll
    for (int nt = 0; nt < 2; ++nt) {
      const int colc = (p * 4 + gp * 2 + nt) * 16 + l16;   // 0..255 in [dw|rw] cat
      const float b1v = b1_lds[colc];
      const float w2v = w2_lds[colc];
      const v8f a = nt ? acc1 : acc0;
      #pragma unroll
      for (int r = 0; r < 8; ++r) {
        float hv = a[r] + b1v;
        float ge = 0.5f * hv * (1.0f + erff(hv * 0.70710678118654752f));
        if (p < 2) pd[r] += ge * w2v; else pr[r] += ge * w2v;
      }
    }
  }

  // reduce layer-2 partials across the 16 lanes of each half-wave
  #pragma unroll
  for (int m = 8; m >= 1; m >>= 1) {
    #pragma unroll
    for (int r = 0; r < 8; ++r) {
      pd[r] += __shfl_xor(pd[r], m, 32);
      pr[r] += __shfl_xor(pr[r], m, 32);
    }
  }
  if (l16 == 0) {
    const int rbase = mt * 16 + hl * 8;
    #pragma unroll
    for (int r = 0; r < 8; ++r) {
      drp[0][rbase + r][gp] = pd[r];
      drp[1][rbase + r][gp] = pr[r];
    }
  }
  __syncthreads();

  // ---- combine halves, EML energy ----
  if (tid < 64) {
    float dr = drp[0][tid][0] + drp[0][tid][1] + db2[0];
    float rs = drp[1][tid][0] + drp[1][tid][1] + rb2[0];
    out_drive[tokBase + tid] = dr;
    out_res[tokBase + tid]   = rs;
    float sp = (rs > 20.f) ? rs : log1pf(expf(rs));   // stable softplus
    float raw = eml_gamma[0] * dr - eml_lambda[0] * sp + eml_bias[0];
    float e = fminf(fmaxf(raw, -3.0f), 3.0f);          // WARMUP_ETA == 1
    out_energy[tokBase + tid] = e;
  }
}

// ---------------------------------------------------------------------------
// k2: softmax over N per batch row.
// ---------------------------------------------------------------------------
__global__ void k2_softmax(const float* __restrict__ energy, float* __restrict__ weights) {
  __shared__ float red[256];
  const int b = blockIdx.x, tid = threadIdx.x;
  const float* e = energy + (long long)b * NN;
  float mx = -1e30f;
  for (int i = tid; i < NN; i += 256) mx = fmaxf(mx, e[i]);
  red[tid] = mx; __syncthreads();
  for (int s = 128; s > 0; s >>= 1) { if (tid < s) red[tid] = fmaxf(red[tid], red[tid + s]); __syncthreads(); }
  mx = red[0]; __syncthreads();
  float sum = 0.f;
  for (int i = tid; i < NN; i += 256) sum += expf(e[i] - mx);
  red[tid] = sum; __syncthreads();
  for (int s = 128; s > 0; s >>= 1) { if (tid < s) red[tid] += red[tid + s]; __syncthreads(); }
  const float inv = 1.0f / red[0];
  for (int i = tid; i < NN; i += 256)
    weights[(long long)b * NN + i] = expf(e[i] - mx) * inv;
}

// ---------------------------------------------------------------------------
// k3a: partial weighted sums  part[b][nc][d] = sum_{n in chunk} w[b,n]*nrm[b,n,d]
// deterministic (no atomics).
// ---------------------------------------------------------------------------
__global__ void k3a_wsum(const float* __restrict__ weights,
                         const unsigned short* __restrict__ nrm,
                         float* __restrict__ part) {
  const int nc = blockIdx.x, b = blockIdx.y, tid = threadIdx.x;
  const int d0 = tid * 2;
  const long long base = (long long)b * NN + nc * 256;
  float a0 = 0.f, a1 = 0.f;
  for (int j = 0; j < 256; ++j) {
    const long long row = base + j;
    if (j + 8 < 256)
      __builtin_prefetch(nrm + (row + 8) * DD + d0, 0, 1);  // -> global_prefetch_b8
    const float w = weights[row];                            // scalar broadcast
    const unsigned pv = *(const unsigned*)(nrm + row * DD + d0);
    a0 += w * bf2f((unsigned short)(pv & 0xffff));
    a1 += w * bf2f((unsigned short)(pv >> 16));
  }
  float* dst = part + ((long long)b * 16 + nc) * DD + d0;
  dst[0] = a0; dst[1] = a1;
}

// ---------------------------------------------------------------------------
// k3b: s = reduce(part);  pooled = s @ vw + vb  via v_wmma_f32_16x16x4_f32.
// One block, 8 waves; M=32 (2 tiles), N=512 (32 tiles), K=512 (128 steps of 4).
// ---------------------------------------------------------------------------
__global__ void __launch_bounds__(256)
k3b_pool(const float* __restrict__ part, const float* __restrict__ vw,
         const float* __restrict__ vb, float* __restrict__ pooled) {
  __shared__ float s_lds[BB * DD];   // 64 KB
  const int tid = threadIdx.x;
  for (int i = tid; i < BB * DD; i += 256) {
    const int b = i / DD, d = i % DD;
    float acc = 0.f;
    #pragma unroll
    for (int c = 0; c < 16; ++c) acc += part[((long long)b * 16 + c) * DD + d];
    s_lds[i] = acc;
  }
  __syncthreads();

  const int wave = tid >> 5, lane = tid & 31;
  const int mt = wave & 1;            // M tile
  const int ng = wave >> 1;           // N tile group (8 each)
  const int l16 = lane & 15, hl = lane >> 4;
  for (int ni = 0; ni < 8; ++ni) {
    const int ntile = ng * 8 + ni;
    const int col = ntile * 16 + l16;
    v8f acc = (v8f){0,0,0,0,0,0,0,0};
    for (int ks = 0; ks < 128; ++ks) {
      const int k = ks * 4 + hl * 2;
      // A 16x4 f32: lane -> M=lane&15, K = (lane>>4)*2 + {0,1}
      v2f Af = (v2f){ s_lds[(mt * 16 + l16) * DD + k],
                      s_lds[(mt * 16 + l16) * DD + k + 1] };
      // B 4x16 f32: lane -> N=lane&15, K = (lane>>4)*2 + {0,1}
      v2f Bf = (v2f){ vw[k * DD + col], vw[(k + 1) * DD + col] };
      acc = __builtin_amdgcn_wmma_f32_16x16x4_f32(
          false, Af, false, Bf, (short)0, acc, false, false);
    }
    const float bias = vb[col];
    #pragma unroll
    for (int r = 0; r < 8; ++r) {
      const int row = mt * 16 + hl * 8 + r;
      pooled[row * DD + col] = acc[r] + bias;
    }
  }
}

// ---------------------------------------------------------------------------
extern "C" void kernel_launch(void* const* d_in, const int* in_sizes, int n_in,
                              void* d_out, int out_size, void* d_ws, size_t ws_size,
                              hipStream_t stream) {
  const float* tokens     = (const float*)d_in[0];
  const float* ln_g       = (const float*)d_in[1];
  const float* ln_b       = (const float*)d_in[2];
  const float* dw1        = (const float*)d_in[3];
  const float* db1        = (const float*)d_in[4];
  const float* dw2        = (const float*)d_in[5];
  const float* db2        = (const float*)d_in[6];
  const float* rw1        = (const float*)d_in[7];
  const float* rb1        = (const float*)d_in[8];
  const float* rw2        = (const float*)d_in[9];
  const float* rb2        = (const float*)d_in[10];
  const float* vw         = (const float*)d_in[11];
  const float* vb         = (const float*)d_in[12];
  const float* eml_gamma  = (const float*)d_in[13];
  const float* eml_lambda = (const float*)d_in[14];
  const float* eml_bias   = (const float*)d_in[15];

  float* out = (float*)d_out;
  float* out_pooled = out;
  float* out_w      = out + OUT_W_OFF;
  float* out_e      = out + OUT_E_OFF;
  float* out_d      = out + OUT_D_OFF;
  float* out_r      = out + OUT_R_OFF;

  char* ws = (char*)d_ws;
  unsigned short* nrm  = (unsigned short*)(ws);
  unsigned short* w1f  = (unsigned short*)(ws + W1F_OFF_B);
  unsigned short* vwf  = (unsigned short*)(ws + VWF_OFF_B);
  float*          part = (float*)(ws + PART_OFF_B);

  k0_prep<<<1024, 256, 0, stream>>>(dw1, rw1, vw, w1f, vwf);
  k1_ln_mlp_energy<<<TOK / 64, 256, 0, stream>>>(
      tokens, ln_g, ln_b, db1, dw2, db2, rb1, rw2, rb2,
      eml_gamma, eml_lambda, eml_bias, w1f, nrm, out_e, out_d, out_r);
  k2_softmax<<<BB, 256, 0, stream>>>(out_e, out_w);
  k3a_wsum<<<dim3(16, BB), 256, 0, stream>>>(out_w, nrm, part);
  k3b_pool<<<1, 256, 0, stream>>>(part, vw, vb, out_pooled);
}
